// biLSTM_44865228374503
// MI455X (gfx1250) — matrix operand order repaired and checked by reading, not compiled, not run on hardware
//
#include <hip/hip_runtime.h>
#include <hip/hip_bf16.h>

typedef __attribute__((ext_vector_type(16))) __bf16 v16bf;
typedef __attribute__((ext_vector_type(8)))  float  v8f;

// Problem constants
constexpr int B_   = 256;
constexpr int SEQ_ = 512;
constexpr int S_   = 511;   // timesteps
constexpr int D_   = 256;
constexpr int H_   = 1024;
constexpr int C_   = 10;
constexpr int KTOT_ = D_ + H_;      // 1280 fused K
constexpr int NKT_  = KTOT_ / 32;   // 40 K-tiles of 32
constexpr int NNT_  = H_ / 16;      // 64 N-tiles of 16 (per gate)
constexpr long PW_ELEMS = 2L * 2 * NNT_ * NKT_ * 512;       // dir,gate,nT,kT,frag
constexpr long PW_BYTES = PW_ELEMS * 2;                     // 10,485,760
constexpr long EMB_BYTES = (long)SEQ_ * D_ * 2;             // 262,144
constexpr long HB_BYTES  = 4L * B_ * H_ * 2;                // [dir][pingpong] bf16
constexpr long CB_BYTES  = 2L * B_ * H_ * 4;                // [dir] f32

// ---------------------------------------------------------------------------
// Prep: convert embedding table to bf16
__global__ void conv_emb_kernel(const float* __restrict__ emb, __bf16* __restrict__ embB) {
    int i = blockIdx.x * 256 + threadIdx.x;
    embB[i] = (__bf16)emb[i];
}

// Prep: pack [Wx;Wh] (K=1280 x N=1024 per gate per dir) into WMMA B-fragment order.
// frag index = ((dir*2+gate)*NNT + nTile)*NKT + kTile ; within frag: lane*16 + j,
// element j of lane L is (K = kTile*32 + (L>=16?16:0) + j, N = nTile*16 + L%16).
__global__ void pack_weights_kernel(
    const float* __restrict__ Wgx_f, const float* __restrict__ Wgh_f,
    const float* __restrict__ Wix_f, const float* __restrict__ Wih_f,
    const float* __restrict__ Wgx_b, const float* __restrict__ Wgh_b,
    const float* __restrict__ Wix_b, const float* __restrict__ Wih_b,
    __bf16* __restrict__ pw)
{
    long idx = (long)blockIdx.x * 256 + threadIdx.x;
    int  r    = (int)(idx & 511);
    long frag = idx >> 9;
    int kTile = (int)(frag % NKT_);
    int nTile = (int)((frag / NKT_) % NNT_);
    int gate  = (int)((frag / ((long)NKT_ * NNT_)) & 1);
    int dir   = (int)(frag / ((long)NKT_ * NNT_ * 2));
    int lane = r >> 4, j = r & 15;
    int K = kTile * 32 + ((lane >= 16) ? 16 : 0) + j;
    int N = nTile * 16 + (lane & 15);
    const float* Wx; const float* Wh;
    if (dir == 0) { Wx = gate ? Wix_f : Wgx_f; Wh = gate ? Wih_f : Wgh_f; }
    else          { Wx = gate ? Wix_b : Wgx_b; Wh = gate ? Wih_b : Wgh_b; }
    float v = (K < D_) ? Wx[(long)K * H_ + N] : Wh[(long)(K - D_) * H_ + N];
    pw[idx] = (__bf16)v;
}

// Prep: zero h ping-pong buffers + c state (re-run every call => deterministic)
__global__ void zero_state_kernel(unsigned int* __restrict__ p, long nWords) {
    long i = (long)blockIdx.x * 256 + threadIdx.x;
    if (i < nWords) p[i] = 0u;
}

// ---------------------------------------------------------------------------
// One LSTM timestep, both directions (blockIdx.z). Fused GEMM:
//   pre_g = [e_t | h] @ [Wgx;Wgh],  pre_i = [e_t | h] @ [Wix;Wih]
// Block: 128 rows x 32 h-cols; 8 waves, wave w owns rows [w*16, w*16+16).
// A tile double-buffered in LDS, filled with gfx1250 async-to-LDS copies.
__global__ __launch_bounds__(256) void lstm_step_kernel(
    const int* __restrict__ x, const __bf16* __restrict__ embB,
    const __bf16* __restrict__ pw,
    const float* __restrict__ bg_f, const float* __restrict__ bi_f,
    const float* __restrict__ bg_b, const float* __restrict__ bi_b,
    __bf16* __restrict__ hBase, float* __restrict__ cBase,
    int t, int bufIn, int bufOut)
{
    __shared__ __bf16 As[2][128 * 32];   // 2 x 8 KB double-buffered A tile

    const int tid  = threadIdx.x;
    const int lane = tid & 31;
    const int wave = tid >> 5;
    const int dir  = blockIdx.z;
    const int mBase = blockIdx.y * 128;
    const int nT0   = blockIdx.x * 2;             // h-col tiles nT0, nT0+1
    const int tTok  = dir ? (S_ - 1 - t) : t;

    const __bf16* hIn  = hBase + (long)(dir * 2 + bufIn)  * (B_ * H_);
    __bf16*       hOut = hBase + (long)(dir * 2 + bufOut) * (B_ * H_);
    float*        cBuf = cBase + (long)dir * (B_ * H_);

    // packed-weight fragment bases: ((dir*2+gate)*NNT + nTile)*NKT*512
    const long fg0 = ((long)(dir * 2 + 0) * NNT_ + nT0)     * NKT_ * 512;
    const long fg1 = ((long)(dir * 2 + 0) * NNT_ + nT0 + 1) * NKT_ * 512;
    const long fi0 = ((long)(dir * 2 + 1) * NNT_ + nT0)     * NKT_ * 512;
    const long fi1 = ((long)(dir * 2 + 1) * NNT_ + nT0 + 1) * NKT_ * 512;
    const long laneOff = (long)lane * 16;

    v8f accG0 = {}, accG1 = {}, accI0 = {}, accI1 = {};

    // A-staging mapping: thread -> (row, 16-elem half); 32B per thread per chunk
    const int rowA  = tid >> 1;
    const int halfA = tid & 1;
    const int bRow  = mBase + rowA;
    const int tok   = x[(long)bRow * SEQ_ + tTok];
    const __bf16* srcEmb = embB + (long)tok * D_;        // K in [0,256)
    const __bf16* srcH   = hIn + (long)bRow * H_ - D_;   // K in [256,1280), add K

    // async copy of one 128x32 A chunk into LDS buffer `buf`
    auto stage_async = [&](int kc, int buf) {
        const int k0 = kc * 32 + halfA * 16;
        const __bf16* src = ((kc < 8) ? srcEmb : srcH) + k0;
        unsigned ldsAddr =
            (unsigned)(uintptr_t)(&As[buf][rowA * 32 + halfA * 16]);
        asm volatile(
            "global_load_async_to_lds_b128 %0, %1, off\n\t"
            "global_load_async_to_lds_b128 %0, %1, off offset:16"
            :: "v"(ldsAddr), "v"(src)
            : "memory");
    };

    // prologue: fill buffer 0
    stage_async(0, 0);
    asm volatile("s_wait_asynccnt 0x0" ::: "memory");
    __syncthreads();

    for (int kc = 0; kc < NKT_; ++kc) {
        const int buf = kc & 1;
        // prefetch next chunk into the other buffer while we compute
        if (kc + 1 < NKT_) stage_async(kc + 1, buf ^ 1);

        // ---- A fragment, contiguous K order (K = 16*(lane>=16) + j), matches packed B ----
        union { v16bf v; unsigned int u[8]; } af;
        const int kb = (lane >= 16) ? 16 : 0;
        const unsigned int* ar = reinterpret_cast<const unsigned int*>(
            &As[buf][(wave * 16 + (lane & 15)) * 32]);
        #pragma unroll
        for (int p = 0; p < 8; ++p) {
            int kk = kb + 2 * p;   // packed-pair K base
            af.u[p] = ar[kk >> 1];
        }

        // ---- B fragments: pre-packed, direct 32B/lane loads (L2-resident) ----
        const long ko = (long)kc * 512 + laneOff;
        v16bf bG0 = *reinterpret_cast<const v16bf*>(pw + fg0 + ko);
        v16bf bG1 = *reinterpret_cast<const v16bf*>(pw + fg1 + ko);
        v16bf bI0 = *reinterpret_cast<const v16bf*>(pw + fi0 + ko);
        v16bf bI1 = *reinterpret_cast<const v16bf*>(pw + fi1 + ko);

        accG0 = __builtin_amdgcn_wmma_f32_16x16x32_bf16(false, af.v, false, bG0, (short)0, accG0, false, false);
        accG1 = __builtin_amdgcn_wmma_f32_16x16x32_bf16(false, af.v, false, bG1, (short)0, accG1, false, false);
        accI0 = __builtin_amdgcn_wmma_f32_16x16x32_bf16(false, af.v, false, bI0, (short)0, accI0, false, false);
        accI1 = __builtin_amdgcn_wmma_f32_16x16x32_bf16(false, af.v, false, bI1, (short)0, accI1, false, false);

        // my async writes (next buffer) must land before anyone reads them;
        // barrier also guarantees current buffer is fully consumed block-wide
        if (kc + 1 < NKT_) asm volatile("s_wait_asynccnt 0x0" ::: "memory");
        __syncthreads();
    }

    // ---- elementwise gate math; C-layout: VGPR k -> M = k + (lane>=16?8:0) ----
    const int mOff = (lane >= 16) ? 8 : 0;
    const int col0 = nT0 * 16 + (lane & 15);
    const int col1 = col0 + 16;
    const float* bg = dir ? bg_b : bg_f;
    const float* bi = dir ? bi_b : bi_f;
    const float bg0v = bg[col0], bg1v = bg[col1];
    const float bi0v = bi[col0], bi1v = bi[col1];

    #pragma unroll
    for (int k = 0; k < 8; ++k) {
        const int brow = mBase + wave * 16 + k + mOff;
        {
            long ci = (long)brow * H_ + col0;
            float g  = tanhf(accG0[k] + bg0v);
            float ig = 1.0f / (1.0f + __expf(-(accI0[k] + bi0v)));
            float cn = ig * (g + cBuf[ci]);      // c = g*i + i*c
            cBuf[ci] = cn;
            hOut[ci] = (__bf16)(tanhf(cn) * ig); // h = tanh(c)*i
        }
        {
            long ci = (long)brow * H_ + col1;
            float g  = tanhf(accG1[k] + bg1v);
            float ig = 1.0f / (1.0f + __expf(-(accI1[k] + bi1v)));
            float cn = ig * (g + cBuf[ci]);
            cBuf[ci] = cn;
            hOut[ci] = (__bf16)(tanhf(cn) * ig);
        }
    }
}

// ---------------------------------------------------------------------------
// Head: out[b,c] = [hf|hb] @ Wp + bp   (tiny: 256x10, plain VALU)
__global__ void head_kernel(const __bf16* __restrict__ hBase,
                            const float* __restrict__ Wp,
                            const float* __restrict__ bp,
                            float* __restrict__ out)
{
    int b = blockIdx.x * 64 + threadIdx.x;
    if (b >= B_) return;
    const int buf = S_ & 1;   // final h lives in ping-pong buffer 1
    const __bf16* hf = hBase + (long)(0 * 2 + buf) * (B_ * H_) + (long)b * H_;
    const __bf16* hb = hBase + (long)(1 * 2 + buf) * (B_ * H_) + (long)b * H_;
    float acc[C_];
    #pragma unroll
    for (int c = 0; c < C_; ++c) acc[c] = bp[c];
    for (int n = 0; n < H_; ++n) {
        float hv = (float)hf[n];
        #pragma unroll
        for (int c = 0; c < C_; ++c) acc[c] += hv * Wp[(long)n * C_ + c];
    }
    for (int n = 0; n < H_; ++n) {
        float hv = (float)hb[n];
        #pragma unroll
        for (int c = 0; c < C_; ++c) acc[c] += hv * Wp[(long)(H_ + n) * C_ + c];
    }
    #pragma unroll
    for (int c = 0; c < C_; ++c) out[(long)b * C_ + c] = acc[c];
}

// ---------------------------------------------------------------------------
extern "C" void kernel_launch(void* const* d_in, const int* in_sizes, int n_in,
                              void* d_out, int out_size, void* d_ws, size_t ws_size,
                              hipStream_t stream) {
    const int*   x     = (const int*)  d_in[0];
    const float* emb   = (const float*)d_in[1];
    const float* Wgx_f = (const float*)d_in[2];
    const float* Wgh_f = (const float*)d_in[3];
    const float* bg_f  = (const float*)d_in[4];
    const float* Wix_f = (const float*)d_in[5];
    const float* Wih_f = (const float*)d_in[6];
    const float* bi_f  = (const float*)d_in[7];
    const float* Wgx_b = (const float*)d_in[8];
    const float* Wgh_b = (const float*)d_in[9];
    const float* bg_b  = (const float*)d_in[10];
    const float* Wix_b = (const float*)d_in[11];
    const float* Wih_b = (const float*)d_in[12];
    const float* bi_b  = (const float*)d_in[13];
    const float* Wp    = (const float*)d_in[14];
    const float* bp    = (const float*)d_in[15];

    char* ws = (char*)d_ws;
    __bf16* pw    = (__bf16*)(ws);
    __bf16* embB  = (__bf16*)(ws + PW_BYTES);
    __bf16* hBase = (__bf16*)(ws + PW_BYTES + EMB_BYTES);
    float*  cBase = (float*) (ws + PW_BYTES + EMB_BYTES + HB_BYTES);

    // Prep (runs every call -> deterministic across graph replays)
    pack_weights_kernel<<<(int)(PW_ELEMS / 256), 256, 0, stream>>>(
        Wgx_f, Wgh_f, Wix_f, Wih_f, Wgx_b, Wgh_b, Wix_b, Wih_b, pw);
    conv_emb_kernel<<<(SEQ_ * D_) / 256, 256, 0, stream>>>(emb, embB);
    const long zWords = (HB_BYTES + CB_BYTES) / 4;
    zero_state_kernel<<<(int)((zWords + 255) / 256), 256, 0, stream>>>(
        (unsigned int*)(ws + PW_BYTES + EMB_BYTES), zWords);

    // Recurrence: 511 steps, fwd+bwd fused per launch
    dim3 grid(H_ / 32, B_ / 128, 2);   // 32 x 2 x 2 = 128 blocks
    for (int t = 0; t < S_; ++t) {
        lstm_step_kernel<<<grid, 256, 0, stream>>>(
            x, embB, pw, bg_f, bi_f, bg_b, bi_b,
            hBase, cBase, t, t & 1, (t + 1) & 1);
    }

    head_kernel<<<(B_ + 63) / 64, 64, 0, stream>>>(hBase, Wp, bp, (float*)d_out);
}
